// GraphAttention_80917183857229
// MI455X (gfx1250) — matrix operand order repaired and checked by reading, not compiled, not run on hardware
//
#include <hip/hip_runtime.h>
#include <hip/hip_bf16.h>

#define NN    8192
#define FF    128
#define NEWF  64
#define SPLIT 8
#define JCHUNK (NN / SPLIT)          // 1024 columns per wave
#define ALPHA 0.2f
#define L2E   1.44269504088896340736f

// LDS slab: 64 rows (features) x 1024 cols bf16, row stride padded to 129 uint4
#define SLAB_STRIDE 129              // uint4 units (128 data + 1 pad)
#define SLAB_BYTES  (64 * SLAB_STRIDE * 16)   // 132096 B

typedef __attribute__((ext_vector_type(16))) __bf16 v16bf;
typedef __attribute__((ext_vector_type(8)))  float  v8f;

#if __has_builtin(__builtin_amdgcn_exp2f)
#define EXP2(x) __builtin_amdgcn_exp2f(x)
#else
#define EXP2(x) exp2f(x)
#endif

// float -> bf16 RNE (used only in the tiny prep kernel)
__device__ __forceinline__ unsigned short f2bf(float f) {
  union { float f; unsigned u; } v; v.f = f;
  unsigned u = v.u;
  unsigned r = u + 0x7fffu + ((u >> 16) & 1u);
  return (unsigned short)(r >> 16);
}

union Pack16 { unsigned u32[8]; uint4 q[2]; v16bf v; };

// ---------------------------------------------------------------------------
// Kernel 1: h = X@W ; store h^T as bf16 ; s = h@a_self ; t = h@a_neigh
// ---------------------------------------------------------------------------
__global__ void gat_prep(const float* __restrict__ X, const float* __restrict__ W,
                         const float* __restrict__ a_self, const float* __restrict__ a_neigh,
                         unsigned short* __restrict__ hbT, float* __restrict__ s,
                         float* __restrict__ t) {
  const int li = threadIdx.x >> 6;        // local row 0..3
  const int f  = threadIdx.x & 63;        // feature 0..63
  const int i  = blockIdx.x * 4 + li;
  const float* xr = X + (size_t)i * FF;
  float dot = 0.f;
#pragma unroll 16
  for (int k = 0; k < FF; ++k) dot += xr[k] * W[k * NEWF + f];
  hbT[(size_t)f * NN + i] = f2bf(dot);    // transposed bf16 copy for WMMA B-operand

  __shared__ float ss[4], st[4];
  if (threadIdx.x < 4) { ss[threadIdx.x] = 0.f; st[threadIdx.x] = 0.f; }
  __syncthreads();
  atomicAdd(&ss[li], dot * a_self[f]);    // ds_add_f32
  atomicAdd(&st[li], dot * a_neigh[f]);
  __syncthreads();
  if (f == 0) { s[i] = ss[li]; t[i] = st[li]; }
}

// ---------------------------------------------------------------------------
// Kernel 2: Mlr = lrelu(max(s) + max(t))  -- global logit upper bound
// ---------------------------------------------------------------------------
__global__ void gat_max(const float* __restrict__ s, const float* __restrict__ t,
                        float* __restrict__ Mlr) {
  __shared__ float rs[256], rt[256];
  float ms = -1e30f, mt = -1e30f;
  for (int idx = threadIdx.x; idx < NN; idx += 256) {
    ms = fmaxf(ms, s[idx]);
    mt = fmaxf(mt, t[idx]);
  }
  rs[threadIdx.x] = ms; rt[threadIdx.x] = mt;
  __syncthreads();
  for (int off = 128; off > 0; off >>= 1) {
    if (threadIdx.x < off) {
      rs[threadIdx.x] = fmaxf(rs[threadIdx.x], rs[threadIdx.x + off]);
      rt[threadIdx.x] = fmaxf(rt[threadIdx.x], rt[threadIdx.x + off]);
    }
    __syncthreads();
  }
  if (threadIdx.x == 0) {
    float x = rs[0] + rt[0];
    *Mlr = x > 0.f ? x : ALPHA * x;       // lrelu is monotone -> valid bound
  }
}

// ---------------------------------------------------------------------------
// Kernel 3: fused mask + exp + (P @ h) partial sums, WMMA bf16.
// Block = 8 waves sharing ONE column split: the 128 KB h^T slab for that
// split is staged once in LDS (padded stride, conflict-free ds_load_b128),
// replacing four serial L2 round-trips per iteration with LDS reads.
// Wave (tile = blkTile*8+wave, sp) covers 16 rows x 1024 columns.
// acc[4] = P @ ones = per-row denominator, same C-layout as outputs.
// ---------------------------------------------------------------------------
__global__ void __launch_bounds__(256, 1) gat_attn(
    const int* __restrict__ A, const unsigned short* __restrict__ hbT,
    const float* __restrict__ s, const float* __restrict__ t,
    const float* __restrict__ Mptr, float* __restrict__ num,
    float* __restrict__ den) {
  extern __shared__ uint4 slab[];             // [64][SLAB_STRIDE] uint4

  const int lane = threadIdx.x & 31;
  const int wave = threadIdx.x >> 5;
  const int sp   = blockIdx.x & 7;            // column split (shared by block)
  const int tile = (blockIdx.x >> 3) * 8 + wave;  // 16-row tile
  const int row  = lane & 15;                 // M index of this lane
  const int half = lane >> 4;                 // K-half select (A-operand layout)
  const int i    = tile * 16 + row;
  const int jbeg = sp * JCHUNK;
  const int jend = jbeg + JCHUNK;

  // ---- stage h^T slab for this split into LDS (coalesced) ----
  for (int idx = threadIdx.x; idx < 64 * 128; idx += 256) {
    const int f = idx >> 7;                   // feature row 0..63
    const int c = idx & 127;                  // 16B chunk within row
    slab[f * SLAB_STRIDE + c] =
        *(const uint4*)(hbT + (size_t)f * NN + jbeg + c * 8);
  }
  __syncthreads();

  const float s_i     = s[i];
  const float negMlog = -(*Mptr) * L2E;       // -M * log2(e)
  const int*  Arow    = A + (size_t)i * NN;

  Pack16 ones;                                // B = all 1.0 bf16 (row-sum trick)
#pragma unroll
  for (int k = 0; k < 8; ++k) ones.u32[k] = 0x3F803F80u;

  v8f acc[5] = {};

  for (int j0 = jbeg; j0 < jend; j0 += 32) {
    const int c0 = j0 + half * 8;             // lane's K group 0 columns
    const int c1 = c0 + 16;                   // lane's K group 1 columns
    if (j0 + 256 < jend) __builtin_prefetch(Arow + c0 + 256, 0, 3);

    // ---- global loads: A mask + t logits (the HBM stream) ----
    int4 a0 = *(const int4*)(Arow + c0);
    int4 a1 = *(const int4*)(Arow + c0 + 4);
    int4 a2 = *(const int4*)(Arow + c1);
    int4 a3 = *(const int4*)(Arow + c1 + 4);
    float4 t0 = *(const float4*)(t + c0);
    float4 t1 = *(const float4*)(t + c0 + 4);
    float4 t2 = *(const float4*)(t + c1);
    float4 t3 = *(const float4*)(t + c1 + 4);

    // ---- B operand tiles from LDS ----
    const int cloc = ((j0 - jbeg) >> 3) + half;       // uint4 index within row
    Pack16 pb[4];
#pragma unroll
    for (int n = 0; n < 4; ++n) {
      const uint4* bp = slab + (n * 16 + row) * SLAB_STRIDE + cloc;
      pb[n].q[0] = bp[0];
      pb[n].q[1] = bp[2];
    }

    const int   av[16] = {a0.x,a0.y,a0.z,a0.w, a1.x,a1.y,a1.z,a1.w,
                          a2.x,a2.y,a2.z,a2.w, a3.x,a3.y,a3.z,a3.w};
    const float tv[16] = {t0.x,t0.y,t0.z,t0.w, t1.x,t1.y,t1.z,t1.w,
                          t2.x,t2.y,t2.z,t2.w, t3.x,t3.y,t3.z,t3.w};

    // Branch-free: mask folded into exponent bias (cndmask of constants).
    float pv[16];
#pragma unroll
    for (int e = 0; e < 16; ++e) {
      float x    = s_i + tv[e];
      float y    = fmaxf(x, ALPHA * x);                 // leaky relu
      float bias = (av[e] != 0) ? negMlog : -1e30f;     // v_cndmask
      pv[e] = EXP2(__builtin_fmaf(y, L2E, bias));       // v_fma + v_exp_f32
    }

    // Pack to bf16 (truncate): one v_perm_b32 per pair.
    Pack16 pa;
#pragma unroll
    for (int k = 0; k < 8; ++k)
      pa.u32[k] = __builtin_amdgcn_perm(__builtin_bit_cast(unsigned, pv[2 * k + 1]),
                                        __builtin_bit_cast(unsigned, pv[2 * k]),
                                        0x07060302u);

    // Row-sum WMMA first: depends only on pa.
    acc[4] = __builtin_amdgcn_wmma_f32_16x16x32_bf16(
        false, pa.v, false, ones.v, (short)0, acc[4], false, false);
#pragma unroll
    for (int n = 0; n < 4; ++n)
      acc[n] = __builtin_amdgcn_wmma_f32_16x16x32_bf16(
          false, pa.v, false, pb[n].v, (short)0, acc[n], false, false);
  }

  // Write unnormalized partials. C-layout: element r = row (half*8 + r).
  const size_t base = (size_t)(tile * SPLIT + sp) * 16;
#pragma unroll
  for (int r = 0; r < 8; ++r) {
    const size_t o = (base + half * 8 + r) * NEWF + row;
#pragma unroll
    for (int n = 0; n < 4; ++n) num[o + n * 16] = acc[n][r];
  }
  if (row == 0) {                              // lanes 0 and 16 cover 16 rows
#pragma unroll
    for (int r = 0; r < 8; ++r) den[base + half * 8 + r] = acc[4][r];
  }
}

// ---------------------------------------------------------------------------
// Kernel 4: deterministic 8-way merge + normalize + relu. Coalesced output.
// ---------------------------------------------------------------------------
__global__ void gat_merge(const float* __restrict__ num, const float* __restrict__ den,
                          float* __restrict__ out) {
  const int idx  = blockIdx.x * 256 + threadIdx.x;   // 0 .. NN*NEWF-1
  const int i    = idx >> 6;
  const int f    = idx & 63;
  const int tile = i >> 4;
  const int r    = i & 15;
  float ns = 0.f, ds = 0.f;
#pragma unroll
  for (int sp = 0; sp < SPLIT; ++sp) {
    const size_t rr = (size_t)(tile * SPLIT + sp) * 16 + r;
    ns += num[rr * NEWF + f];
    ds += den[rr];
  }
  float v = ds > 0.f ? ns / ds : 0.f;
  out[idx] = v > 0.f ? v : 0.f;                      // final relu
}

// ---------------------------------------------------------------------------
extern "C" void kernel_launch(void* const* d_in, const int* in_sizes, int n_in,
                              void* d_out, int out_size, void* d_ws, size_t ws_size,
                              hipStream_t stream) {
  const float* X       = (const float*)d_in[0];
  const int*   A       = (const int*)  d_in[1];
  const float* W       = (const float*)d_in[2];
  const float* a_self  = (const float*)d_in[3];
  const float* a_neigh = (const float*)d_in[4];
  float* out = (float*)d_out;

  char* ws = (char*)d_ws;
  size_t off = 0;
  unsigned short* hbT = (unsigned short*)(ws + off); off += (size_t)NEWF * NN * 2;  // 1 MB
  float* s   = (float*)(ws + off); off += (size_t)NN * 4;
  float* t   = (float*)(ws + off); off += (size_t)NN * 4;
  float* Mlr = (float*)(ws + off); off += 64;                       // padded
  float* num = (float*)(ws + off); off += (size_t)(NN / 16) * SPLIT * 16 * NEWF * 4; // 16.8 MB
  float* den = (float*)(ws + off);                                  // 0.26 MB

  gat_prep <<<NN / 4, 256, 0, stream>>>(X, W, a_self, a_neigh, hbT, s, t);
  gat_max  <<<1,      256, 0, stream>>>(s, t, Mlr);
  gat_attn <<<(NN / 16) * SPLIT / 8, 256, SLAB_BYTES, stream>>>(A, hbT, s, t, Mlr, num, den);
  gat_merge<<<NN * NEWF / 256, 256, 0, stream>>>(num, den, out);
}
